// MILNCE_naive_61091614818453
// MI455X (gfx1250) — matrix operand order repaired and checked by reading, not compiled, not run on hardware
//
#include <hip/hip_runtime.h>
#include <hip/hip_bf16.h>
#include <math.h>

typedef __attribute__((ext_vector_type(16))) _Float16 v16h;
typedef __attribute__((ext_vector_type(8)))  _Float16 v8h;
typedef __attribute__((ext_vector_type(8)))  float    v8f;

#define N_TOK   2048
#define NROW    6144          // 3 * 2048
#define DDIM    256
#define TAU_INV 10.0f

union FragU { v16h v; v8h p[2]; };

// ---------------------------------------------------------------------------
// Kernel 1: per-row L2 norm + normalize + convert to f16 (H in workspace)
// ---------------------------------------------------------------------------
__global__ __launch_bounds__(256) void milnce_normalize(
    const float* __restrict__ a, const float* __restrict__ s2,
    const float* __restrict__ s1, _Float16* __restrict__ H)
{
    __shared__ float red[256];
    const int r = blockIdx.x, tid = threadIdx.x;
    const float* src = (r < N_TOK)     ? (a  + (size_t)r            * DDIM)
                     : (r < 2 * N_TOK) ? (s2 + (size_t)(r - N_TOK)  * DDIM)
                                       : (s1 + (size_t)(r - 2*N_TOK)* DDIM);
    const float x = src[tid];
    red[tid] = x * x;
    __syncthreads();
    #pragma unroll
    for (int s = 128; s > 0; s >>= 1) {
        if (tid < s) red[tid] += red[tid + s];
        __syncthreads();
    }
    // reference clips norm_i*norm_j at 1e-8; clamping each norm at 1e-4 matches
    const float inv = 1.0f / fmaxf(sqrtf(red[0]), 1e-4f);
    H[(size_t)r * DDIM + tid] = (_Float16)(x * inv);
}

// ---------------------------------------------------------------------------
// Kernel 2: logits = H * H^T via v_wmma_f32_16x16x32_f16.
// 8 waves / block; wave -> 32x32 patch (2x2 tiles); block tile 128(y) x 64(x).
// A frag: lane L -> row m_base+(L&15); halves at k0+(L>>4)*8 and +16 (2x16B).
// B frag: B=H^T so B[k][n]=H[n_base+n][k]; lane L -> row n_base+(L&15),
//         halves at k0+(L>>4)*8 and +16 (2x16B).
// ---------------------------------------------------------------------------
__global__ __launch_bounds__(256) void milnce_gemm(
    const _Float16* __restrict__ H, float* __restrict__ C)
{
    const int lane = threadIdx.x & 31;
    const int wave = threadIdx.x >> 5;
    const int wm = wave & 3, wn = wave >> 2;
    const int m_base = blockIdx.y * 128 + wm * 32;
    const int n_base = blockIdx.x * 64  + wn * 32;
    const int lrow = lane & 15, lhi = lane >> 4;

    const _Float16* aRow0 = H + (size_t)(m_base + lrow) * DDIM;
    const _Float16* aRow1 = aRow0 + (size_t)16 * DDIM;
    const _Float16* bRow0 = H + (size_t)(n_base + lrow) * DDIM;
    const _Float16* bRow1 = bRow0 + (size_t)16 * DDIM;

    v8f acc00 = {}, acc01 = {}, acc10 = {}, acc11 = {};

    #pragma unroll
    for (int k0 = 0; k0 < DDIM; k0 += 32) {
        FragU a0, a1, b0, b1;
        const int ka = k0 + lhi * 8;     // A: K blocks {0..7,16..23} / {8..15,24..31}
        a0.p[0] = *(const v8h*)(aRow0 + ka);
        a0.p[1] = *(const v8h*)(aRow0 + ka + 16);
        a1.p[0] = *(const v8h*)(aRow1 + ka);
        a1.p[1] = *(const v8h*)(aRow1 + ka + 16);
        const int kb = k0 + lhi * 8;     // B: K blocks {0..7,16..23} / {8..15,24..31}
        b0.p[0] = *(const v8h*)(bRow0 + kb);
        b0.p[1] = *(const v8h*)(bRow0 + kb + 16);
        b1.p[0] = *(const v8h*)(bRow1 + kb);
        b1.p[1] = *(const v8h*)(bRow1 + kb + 16);

        acc00 = __builtin_amdgcn_wmma_f32_16x16x32_f16(false, a0.v, false, b0.v,
                                                       (short)0, acc00, false, false);
        acc01 = __builtin_amdgcn_wmma_f32_16x16x32_f16(false, a0.v, false, b1.v,
                                                       (short)0, acc01, false, false);
        acc10 = __builtin_amdgcn_wmma_f32_16x16x32_f16(false, a1.v, false, b0.v,
                                                       (short)0, acc10, false, false);
        acc11 = __builtin_amdgcn_wmma_f32_16x16x32_f16(false, a1.v, false, b1.v,
                                                       (short)0, acc11, false, false);
    }

    // C/D layout: VGPR r, lanes 0-15 -> M=r, lanes 16-31 -> M=8+r; N = lane&15
    v8f accs[2][2] = { { acc00, acc01 }, { acc10, acc11 } };
    #pragma unroll
    for (int i = 0; i < 2; ++i) {
        #pragma unroll
        for (int j = 0; j < 2; ++j) {
            const int col = n_base + 16 * j + lrow;
            #pragma unroll
            for (int rr = 0; rr < 8; ++rr) {
                const int row = m_base + 16 * i + lhi * 8 + rr;
                C[(size_t)row * NROW + col] = accs[i][j][rr];
            }
        }
    }
}

// ---------------------------------------------------------------------------
// Kernel 3: per-row loss contribution
//   lse_offdiag(logits[r,:]/tau) - lse(pos/tau), pos = same token, other mods
// ---------------------------------------------------------------------------
__global__ __launch_bounds__(256) void milnce_loss(
    const float* __restrict__ logits, float* __restrict__ contrib)
{
    __shared__ float red[256];
    const int r = blockIdx.x, tid = threadIdx.x;
    const float* row = logits + (size_t)r * NROW;

    // pass 1: max over off-diagonal scaled logits
    float lmax = -INFINITY;
    for (int j = tid; j < NROW; j += 256)
        if (j != r) lmax = fmaxf(lmax, row[j] * TAU_INV);
    red[tid] = lmax;
    __syncthreads();
    #pragma unroll
    for (int s = 128; s > 0; s >>= 1) {
        if (tid < s) red[tid] = fmaxf(red[tid], red[tid + s]);
        __syncthreads();
    }
    const float m = red[0];
    __syncthreads();

    // pass 2: sum of exp
    float ssum = 0.0f;
    for (int j = tid; j < NROW; j += 256)
        if (j != r) ssum += expf(row[j] * TAU_INV - m);
    red[tid] = ssum;
    __syncthreads();
    #pragma unroll
    for (int s = 128; s > 0; s >>= 1) {
        if (tid < s) red[tid] += red[tid + s];
        __syncthreads();
    }

    if (tid == 0) {
        const float lse = m + logf(red[0]);
        const int k = r >> 11;          // modality
        const int i = r & (N_TOK - 1);  // token index
        float p[2]; int c = 0;
        #pragma unroll
        for (int jm = 0; jm < 3; ++jm)
            if (jm != k) p[c++] = row[i + (jm << 11)] * TAU_INV;
        const float pm   = fmaxf(p[0], p[1]);
        const float plse = pm + logf(expf(p[0] - pm) + expf(p[1] - pm));
        contrib[r] = lse - plse;
    }
}

// ---------------------------------------------------------------------------
// Kernel 4: deterministic fixed-order reduction -> loss / N
// ---------------------------------------------------------------------------
__global__ __launch_bounds__(256) void milnce_final(
    const float* __restrict__ contrib, float* __restrict__ out)
{
    __shared__ float red[256];
    const int tid = threadIdx.x;
    float s = 0.0f;
    for (int r = tid; r < NROW; r += 256) s += contrib[r];
    red[tid] = s;
    __syncthreads();
    #pragma unroll
    for (int st = 128; st > 0; st >>= 1) {
        if (tid < st) red[tid] += red[tid + st];
        __syncthreads();
    }
    if (tid == 0) out[0] = red[0] / (float)NROW;
}

// ---------------------------------------------------------------------------
extern "C" void kernel_launch(void* const* d_in, const int* in_sizes, int n_in,
                              void* d_out, int out_size, void* d_ws, size_t ws_size,
                              hipStream_t stream)
{
    const float* a  = (const float*)d_in[0];
    const float* s2 = (const float*)d_in[1];
    const float* s1 = (const float*)d_in[2];
    float* out    = (float*)d_out;
    float* logits = out + 1;   // d_out layout: [loss][logits row-major]

    _Float16* H      = (_Float16*)d_ws;                               // 3 MB
    float*    contrib = (float*)((char*)d_ws +
                          (size_t)NROW * DDIM * sizeof(_Float16));     // 24 KB

    milnce_normalize<<<NROW, 256, 0, stream>>>(a, s2, s1, H);

    dim3 grid(NROW / 64, NROW / 128);
    milnce_gemm<<<grid, 256, 0, stream>>>(H, logits);

    milnce_loss<<<NROW, 256, 0, stream>>>(logits, contrib);
    milnce_final<<<1, 256, 0, stream>>>(contrib, out);
}
